// S4D_29480655519988
// MI455X (gfx1250) — matrix operand order repaired and checked by reading, not compiled, not run on hardware
//
#include <hip/hip_runtime.h>
#include <hip/hip_bf16.h>
#include <hip/hip_fp16.h>

typedef __attribute__((ext_vector_type(16))) _Float16 v16h;
typedef __attribute__((ext_vector_type(8)))  float    v8f;

#define Hn   256
#define N2n  32
#define Bn   16
#define Ln   8192
#define Tc   64            // chunk length
#define NSEG 8             // parallel segments over L
#define LS   (Ln / NSEG)   // 1024 elements per segment
#define NCHS (LS / Tc)     // 16 chunks per segment

// ---------------------------------------------------------------------------
// Setup: per h, build the three 64x64 f16 operand matrices + decay factors.
//   Tk[t][s]   = k[t-s] (lower-tri Toeplitz of kernel taps)
//   Pm[t][2n]  =  2*Re(Ct_n * w_n^{t+1});  Pm[t][2n+1] = -2*Im(Ct_n * w_n^{t+1})
//   Qm[2n][s]  =  Re(w_n^{Tc-1-s});        Qm[2n+1][s] =  Im(w_n^{Tc-1-s})
//   wT[n]      =  w_n^Tc   (per-chunk decay, f32)
//   wLS[n]     =  w_n^LS   (per-segment decay, f32)
// State rows interleaved: row 2n = Re x_n, row 2n+1 = Im x_n.
// ---------------------------------------------------------------------------
__global__ void s4d_setup(const float* __restrict__ log_dt,
                          const float* __restrict__ log_A_real,
                          const float* __restrict__ A_imag,
                          const float* __restrict__ Cmat,
                          _Float16* __restrict__ Tk,
                          _Float16* __restrict__ Pm,
                          _Float16* __restrict__ Qm,
                          float* __restrict__ wTre,  float* __restrict__ wTim,
                          float* __restrict__ wLSre, float* __restrict__ wLSim) {
  const int h = blockIdx.x;
  const int t = threadIdx.x;           // 0..63
  __shared__ float kl[Tc];

  const float dt = __expf(log_dt[h]);

  float acc = 0.f;
  for (int n = 0; n < N2n; ++n) {
    const float rA = -__expf(log_A_real[h * N2n + n]);
    const float iA = A_imag[h * N2n + n];
    const float cr = Cmat[(h * N2n + n) * 2 + 0];
    const float ci = Cmat[(h * N2n + n) * 2 + 1];
    float s1, c1;
    const float er = __expf(rA * dt);
    __sincosf(iA * dt, &s1, &c1);
    const float w1r = er * c1, w1i = er * s1;
    const float inv = 1.0f / (rA * rA + iA * iA);
    const float gr = ((w1r - 1.f) * rA + w1i * iA) * inv;
    const float gi = (w1i * rA - (w1r - 1.f) * iA) * inv;
    const float Ctr = cr * gr - ci * gi;
    const float Cti = cr * gi + ci * gr;
    float ss, cc;
    const float mt = __expf(rA * dt * (float)t);
    __sincosf(iA * dt * (float)t, &ss, &cc);
    acc += Ctr * (mt * cc) - Cti * (mt * ss);
    const float m2 = __expf(rA * dt * (float)(t + 1));
    __sincosf(iA * dt * (float)(t + 1), &ss, &cc);
    const float w2r = m2 * cc, w2i = m2 * ss;
    const float pr = Ctr * w2r - Cti * w2i;
    const float pi = Ctr * w2i + Cti * w2r;
    Pm[(h * Tc + t) * Tc + 2 * n    ] = (_Float16)( 2.f * pr);
    Pm[(h * Tc + t) * Tc + 2 * n + 1] = (_Float16)(-2.f * pi);
  }
  kl[t] = 2.f * acc;

  {
    const int n = t >> 1;
    const float rA = -__expf(log_A_real[h * N2n + n]);
    const float iA = A_imag[h * N2n + n];
    for (int s = 0; s < Tc; ++s) {
      const float e = (float)(Tc - 1 - s);
      float ss, cc;
      const float m = __expf(rA * dt * e);
      __sincosf(iA * dt * e, &ss, &cc);
      Qm[(h * Tc + t) * Tc + s] = (_Float16)((t & 1) ? (m * ss) : (m * cc));
    }
  }
  if (t < N2n) {
    const float rA = -__expf(log_A_real[h * N2n + t]);
    const float iA = A_imag[h * N2n + t];
    float ss, cc;
    float m = __expf(rA * dt * (float)Tc);
    __sincosf(iA * dt * (float)Tc, &ss, &cc);
    wTre[h * N2n + t] = m * cc;
    wTim[h * N2n + t] = m * ss;
    m = __expf(rA * dt * (float)LS);
    __sincosf(iA * dt * (float)LS, &ss, &cc);
    wLSre[h * N2n + t] = m * cc;
    wLSim[h * N2n + t] = m * ss;
  }

  __syncthreads();
  for (int s = 0; s < Tc; ++s)
    Tk[(h * Tc + t) * Tc + s] = (_Float16)((t >= s) ? kl[t - s] : 0.f);
}

// ---------------------------------------------------------------------------
// Pass 1: per (h, segment), run the state path only (zero initial carry) and
// emit the segment summary S[seg] (64 interleaved Re/Im rows x 16 batch cols).
// Per chunk: complex decay (VALU) + 2 WMMAs. u staged once in double-buffered
// LDS (f16), one barrier per chunk.
// ---------------------------------------------------------------------------
__global__ void __launch_bounds__(128)
s4d_pass1(const float* __restrict__ u,
          const _Float16* __restrict__ Qm,
          const float* __restrict__ wTre,
          const float* __restrict__ wTim,
          float* __restrict__ Xseg) {
  const int h     = blockIdx.x;
  const int seg   = blockIdx.y;
  const int tid   = threadIdx.x;
  const int wave  = tid >> 5;
  const int lane  = tid & 31;
  const int lhalf = lane >> 4;
  const int lmod  = lane & 15;

  __shared__ __attribute__((aligned(32))) _Float16 uh[2][Bn * Tc];

  // invariant A-operand fragments of Qm (contiguous k: element i of lane half h is k = 16h + i per 32-k block)
  v16h qmf[2];
  {
    const int mrow = wave * 16 + lmod;
    const _Float16* qmp = Qm + ((size_t)h * Tc + mrow) * Tc;
#pragma unroll
    for (int kk = 0; kk < 2; ++kk) {
      const int kb = kk * 32 + lhalf * 16;
#pragma unroll
      for (int i = 0; i < 16; i += 2) {
        const int k = kb + i;
        qmf[kk][i] = qmp[k]; qmf[kk][i + 1] = qmp[k + 1];
      }
    }
  }
  float dr[4], di[4];
#pragma unroll
  for (int p = 0; p < 4; ++p) {
    const int n = wave * 8 + p + lhalf * 4;
    dr[p] = wTre[h * N2n + n];
    di[p] = wTim[h * N2n + n];
  }

  const v8f zf = {0.f, 0.f, 0.f, 0.f, 0.f, 0.f, 0.f, 0.f};
  v8f xacc = zf;

  const int b0 = tid >> 3;
  const int s0 = (tid & 7) * 8;
  const float* ubase = u + ((size_t)b0 * Hn + h) * Ln + (size_t)seg * LS;

  float ru[8];
#pragma unroll
  for (int q = 0; q < 8; ++q) ru[q] = ubase[s0 + q];

  for (int c = 0; c < NCHS; ++c) {
    const int buf = c & 1;
#pragma unroll
    for (int q = 0; q < 8; ++q) uh[buf][b0 * Tc + s0 + q] = (_Float16)ru[q];
    __syncthreads();
    if (c + 1 < NCHS) {
#pragma unroll
      for (int q = 0; q < 8; ++q) ru[q] = ubase[(c + 1) * Tc + s0 + q];
    }
    v16h bu[2];
#pragma unroll
    for (int kk = 0; kk < 2; ++kk)
      bu[kk] = *(const v16h*)&uh[buf][lmod * Tc + kk * 32 + lhalf * 16];
#pragma unroll
    for (int p = 0; p < 4; ++p) {
      const float xr = xacc[2 * p], xi = xacc[2 * p + 1];
      xacc[2 * p]     = dr[p] * xr - di[p] * xi;
      xacc[2 * p + 1] = dr[p] * xi + di[p] * xr;
    }
    xacc = __builtin_amdgcn_wmma_f32_16x16x32_f16(false, qmf[0], false, bu[0], (short)0, xacc, false, false);
    xacc = __builtin_amdgcn_wmma_f32_16x16x32_f16(false, qmf[1], false, bu[1], (short)0, xacc, false, false);
  }
  // emit segment summary
#pragma unroll
  for (int r = 0; r < 8; ++r) {
    const int j = wave * 16 + r + lhalf * 8;
    Xseg[(((size_t)h * NSEG + seg) * Tc + j) * Bn + lmod] = xacc[r];
  }
}

// ---------------------------------------------------------------------------
// Pass 2: tiny sequential scan over segment summaries -> per-segment incoming
// carries. One block per h, one thread per (mode n, batch b).
// ---------------------------------------------------------------------------
__global__ void s4d_pass2(const float* __restrict__ Xseg,
                          const float* __restrict__ wLSre,
                          const float* __restrict__ wLSim,
                          float* __restrict__ Xin) {
  const int h = blockIdx.x;
  const int n = threadIdx.x >> 4;   // 0..31
  const int b = threadIdx.x & 15;
  const float ar = wLSre[h * N2n + n];
  const float ai = wLSim[h * N2n + n];
  float cr = 0.f, ci = 0.f;
  for (int seg = 0; seg < NSEG; ++seg) {
    const size_t base = ((size_t)h * NSEG + seg) * Tc;
    Xin[(base + 2 * n    ) * Bn + b] = cr;
    Xin[(base + 2 * n + 1) * Bn + b] = ci;
    const float sr = Xseg[(base + 2 * n    ) * Bn + b];
    const float si = Xseg[(base + 2 * n + 1) * Bn + b];
    const float nr = ar * cr - ai * ci + sr;
    const float ni = ar * ci + ai * cr + si;
    cr = nr; ci = ni;
  }
}

// ---------------------------------------------------------------------------
// Pass 3: per (h, segment), full chunk scan seeded with the incoming carry,
// producing y. Per chunk per wave: y = Tk@U + Pm@X + D*U (4 WMMAs),
// X = decay*X + Qm@U (2 WMMAs, f32 accumulators).
// ---------------------------------------------------------------------------
__global__ void __launch_bounds__(128)
s4d_pass3(const float* __restrict__ u,
          const float* __restrict__ Dv,
          const _Float16* __restrict__ Tk,
          const _Float16* __restrict__ Pm,
          const _Float16* __restrict__ Qm,
          const float* __restrict__ wTre,
          const float* __restrict__ wTim,
          const float* __restrict__ Xin,
          float* __restrict__ y) {
  const int h     = blockIdx.x;
  const int seg   = blockIdx.y;
  const int tid   = threadIdx.x;
  const int wave  = tid >> 5;
  const int lane  = tid & 31;
  const int lhalf = lane >> 4;
  const int lmod  = lane & 15;

  __shared__ __attribute__((aligned(32))) _Float16 uh[Bn * Tc]; // [b][s] f16
  __shared__ __attribute__((aligned(32))) float    uf[Bn * Tc]; // [b][s] f32
  __shared__ __attribute__((aligned(32))) _Float16 xh[Bn * Tc]; // [b][j] state
  __shared__ __attribute__((aligned(32))) float    yl[Bn * Tc]; // [b][t]

  v16h tkf[2], pmf[2], qmf[2];
  {
    const int mrow = wave * 16 + lmod;
    const _Float16* tkp = Tk + ((size_t)h * Tc + mrow) * Tc;
    const _Float16* pmp = Pm + ((size_t)h * Tc + mrow) * Tc;
    const _Float16* qmp = Qm + ((size_t)h * Tc + mrow) * Tc;
#pragma unroll
    for (int kk = 0; kk < 2; ++kk) {
      const int kb = kk * 32 + lhalf * 16;
#pragma unroll
      for (int i = 0; i < 16; i += 2) {
        const int k = kb + i;
        tkf[kk][i] = tkp[k]; tkf[kk][i + 1] = tkp[k + 1];
        pmf[kk][i] = pmp[k]; pmf[kk][i + 1] = pmp[k + 1];
        qmf[kk][i] = qmp[k]; qmf[kk][i + 1] = qmp[k + 1];
      }
    }
  }
  float dr[4], di[4];
#pragma unroll
  for (int p = 0; p < 4; ++p) {
    const int n = wave * 8 + p + lhalf * 4;
    dr[p] = wTre[h * N2n + n];
    di[p] = wTim[h * N2n + n];
  }
  const float Dh = Dv[h];

  // seed state from incoming carry (f32 accumulators + f16 LDS copy)
  v8f xacc;
#pragma unroll
  for (int r = 0; r < 8; ++r) {
    const int j = wave * 16 + r + lhalf * 8;
    const float v = Xin[(((size_t)h * NSEG + seg) * Tc + j) * Bn + lmod];
    xacc[r] = v;
    xh[lmod * Tc + j] = (_Float16)v;
  }

  const int b0 = tid >> 3;
  const int s0 = (tid & 7) * 8;
  const float* ubase = u + ((size_t)b0 * Hn + h) * Ln + (size_t)seg * LS;
  float*       ybase = y + ((size_t)b0 * Hn + h) * Ln + (size_t)seg * LS;

  float ru[8];
#pragma unroll
  for (int q = 0; q < 8; ++q) ru[q] = ubase[s0 + q];

  const v8f zf = {0.f, 0.f, 0.f, 0.f, 0.f, 0.f, 0.f, 0.f};

  for (int c = 0; c < NCHS; ++c) {
#pragma unroll
    for (int q = 0; q < 8; ++q) {
      uf[b0 * Tc + s0 + q] = ru[q];
      uh[b0 * Tc + s0 + q] = (_Float16)ru[q];
    }
    __syncthreads();  // barrier 1: u + state visible

    if (c + 1 < NCHS) {
#pragma unroll
      for (int q = 0; q < 8; ++q) ru[q] = ubase[(c + 1) * Tc + s0 + q];
    }

    v16h bu[2], bx[2];
#pragma unroll
    for (int kk = 0; kk < 2; ++kk) {
      const int off = lmod * Tc + kk * 32 + lhalf * 16;
      bu[kk] = *(const v16h*)&uh[off];
      bx[kk] = *(const v16h*)&xh[off];
    }

    v8f yacc = zf;
    yacc = __builtin_amdgcn_wmma_f32_16x16x32_f16(false, tkf[0], false, bu[0], (short)0, yacc, false, false);
    yacc = __builtin_amdgcn_wmma_f32_16x16x32_f16(false, tkf[1], false, bu[1], (short)0, yacc, false, false);
    yacc = __builtin_amdgcn_wmma_f32_16x16x32_f16(false, pmf[0], false, bx[0], (short)0, yacc, false, false);
    yacc = __builtin_amdgcn_wmma_f32_16x16x32_f16(false, pmf[1], false, bx[1], (short)0, yacc, false, false);

#pragma unroll
    for (int p = 0; p < 4; ++p) {
      const float xr = xacc[2 * p], xi = xacc[2 * p + 1];
      xacc[2 * p]     = dr[p] * xr - di[p] * xi;
      xacc[2 * p + 1] = dr[p] * xi + di[p] * xr;
    }
    xacc = __builtin_amdgcn_wmma_f32_16x16x32_f16(false, qmf[0], false, bu[0], (short)0, xacc, false, false);
    xacc = __builtin_amdgcn_wmma_f32_16x16x32_f16(false, qmf[1], false, bu[1], (short)0, xacc, false, false);

#pragma unroll
    for (int r = 0; r < 8; ++r) {
      const int trow = wave * 16 + r + lhalf * 8;
      yl[lmod * Tc + trow] = yacc[r] + Dh * uf[lmod * Tc + trow];
    }
    __syncthreads();  // barrier 2: reads of uh/xh done, yl complete

#pragma unroll
    for (int r = 0; r < 8; ++r) {
      const int j = wave * 16 + r + lhalf * 8;
      xh[lmod * Tc + j] = (_Float16)xacc[r];
    }
#pragma unroll
    for (int q = 0; q < 8; ++q)
      ybase[c * Tc + s0 + q] = yl[b0 * Tc + s0 + q];
  }
}

// ---------------------------------------------------------------------------
extern "C" void kernel_launch(void* const* d_in, const int* in_sizes, int n_in,
                              void* d_out, int out_size, void* d_ws, size_t ws_size,
                              hipStream_t stream) {
  const float* u          = (const float*)d_in[0];
  const float* log_dt     = (const float*)d_in[1];
  const float* log_A_real = (const float*)d_in[2];
  const float* A_imag     = (const float*)d_in[3];
  const float* C          = (const float*)d_in[4];
  const float* D          = (const float*)d_in[5];
  float* y = (float*)d_out;

  char* ws = (char*)d_ws;
  const size_t MAT  = (size_t)Hn * Tc * Tc * sizeof(_Float16);   // 2 MB each
  const size_t VEC  = (size_t)Hn * N2n * sizeof(float);          // 32 KB each
  const size_t SEGA = (size_t)Hn * NSEG * Tc * Bn * sizeof(float); // 8 MB each
  _Float16* Tk = (_Float16*)(ws);
  _Float16* Pm = (_Float16*)(ws + MAT);
  _Float16* Qm = (_Float16*)(ws + 2 * MAT);
  float* wTre  = (float*)(ws + 3 * MAT);
  float* wTim  = (float*)(ws + 3 * MAT + VEC);
  float* wLSre = (float*)(ws + 3 * MAT + 2 * VEC);
  float* wLSim = (float*)(ws + 3 * MAT + 3 * VEC);
  float* Xseg  = (float*)(ws + 8 * 1024 * 1024);
  float* Xin   = (float*)(ws + 8 * 1024 * 1024 + SEGA);

  s4d_setup<<<dim3(Hn), dim3(Tc), 0, stream>>>(log_dt, log_A_real, A_imag, C,
                                               Tk, Pm, Qm, wTre, wTim, wLSre, wLSim);
  s4d_pass1<<<dim3(Hn, NSEG), dim3(128), 0, stream>>>(u, Qm, wTre, wTim, Xseg);
  s4d_pass2<<<dim3(Hn), dim3(N2n * Bn), 0, stream>>>(Xseg, wLSre, wLSim, Xin);
  s4d_pass3<<<dim3(Hn, NSEG), dim3(128), 0, stream>>>(u, D, Tk, Pm, Qm, wTre, wTim, Xin, y);
}
